// imdct_21440476741756
// MI455X (gfx1250) — matrix operand order repaired and checked by reading, not compiled, not run on hardware
//
#include <hip/hip_runtime.h>

typedef _Float16 v8h  __attribute__((ext_vector_type(8)));
typedef _Float16 v16h __attribute__((ext_vector_type(16)));
typedef float    v8f  __attribute__((ext_vector_type(8)));

#define T_FRAMES 4096
#define N_FREQ   256
#define HOP      256
#define NFFT     512
#define KTOT     512       // combined reduction length (2*N_FREQ)
#define KC       32        // K per v_wmma_f32_16x16x32_f16
#define LOUT     1048832   // (T_FRAMES+1)*HOP == (T-1)*HOP + NFFT
#define LDS_PAD  48        // row stride in halfwords: 96B (16B-aligned, bank-spread)

// out[b, t*HOP + r] = (4/NFFT) * sum_j A'[t,j] * B'[j,r]
//   A'[t,j] = spec[b, j, t]        (j <  256)
//           = spec[b, j-256, t-1]  (j >= 256)   (zero outside [0,T))
//   B'[j,r] = kernels[r, j]        (j <  256)
//           = kernels[r+256,j-256] (j >= 256)
// Disjoint writes (no atomics): every (t in [0,4096], r in [0,256)) written once.
// Workgroup tile: 64 (t) x 128 (r); wave tile: 32 x 32 (4 WMMA accumulators).

struct PrefA { float a[8]; };       // 8 k-values at one t-row (one uniform guard)
struct PrefB { float4 b[4]; };      // 16 consecutive k at one r

__global__ __launch_bounds__(256) void imdct_wmma_kernel(
    const float* __restrict__ spec,     // [16][256][4096]
    const float* __restrict__ kernels,  // [512][256]
    float* __restrict__ out)            // [16][LOUT]
{
  __shared__ __align__(16) _Float16 As[2][64][LDS_PAD];   // [buf][t-row][k]
  __shared__ __align__(16) _Float16 Bs[2][128][LDS_PAD];  // [buf][r][k]

  const int tid  = threadIdx.x;
  const int lane = tid & 31;
  const int wave = tid >> 5;
  const int hf   = lane >> 4;           // half-wave select (WMMA layouts)
  const int ln16 = lane & 15;

  const int m_sub = wave >> 2;          // 0..1 : 32-row sub-tile in t
  const int n_sub = wave & 3;           // 0..3 : 32-col sub-tile in r

  const int r0 = blockIdx.x * 128;      // output column block (r)
  const int t0 = blockIdx.y * 64;       // output row block (t): 0..4096
  const int b  = blockIdx.z;

  // A staging: thread = one t-row, 8 k-values -> ONE uniform bounds check,
  // lane-contiguous in t at fixed f (coalesced 128B per wave per k).
  const int a_i  = tid & 63;            // t row within tile (0..63)
  const int a_j8 = (tid >> 6) * 8;      // k base within chunk: 0,8,16,24
  // B staging: thread = one r, 16 consecutive k (vectorizes to float4 loads)
  const int b_r  = tid >> 1;            // 0..127
  const int b_jh = (tid & 1) * 16;      // 0 or 16

  const float* spec_b = spec + (size_t)b * N_FREQ * T_FRAMES;

  // ---- prefetch (global -> regs) for chunk kk ----
  auto prefetchA = [&](int kk) -> PrefA {
    const int second = (kk >= N_FREQ) ? 1 : 0;
    const int kkm    = kk & (N_FREQ - 1);
    const int tt     = t0 + a_i - second;
    PrefA r;
    #pragma unroll
    for (int u = 0; u < 8; ++u) r.a[u] = 0.f;
    if ((unsigned)tt < (unsigned)T_FRAMES) {     // one predicate, 8 loads
      const float* sp = spec_b + (size_t)(kkm + a_j8) * T_FRAMES + tt;
      #pragma unroll
      for (int u = 0; u < 8; ++u) r.a[u] = sp[(size_t)u * T_FRAMES];
    }
    return r;
  };
  auto prefetchB = [&](int kk) -> PrefB {
    const int second = (kk >= N_FREQ) ? 1 : 0;
    const int kkm    = kk & (N_FREQ - 1);
    const float4* kp = (const float4*)(kernels +
        (size_t)(r0 + b_r + second * HOP) * N_FREQ + kkm + b_jh);
    PrefB r;
    r.b[0] = kp[0]; r.b[1] = kp[1]; r.b[2] = kp[2]; r.b[3] = kp[3];
    return r;
  };
  // ---- commit (regs -> f16 LDS tile) ----
  auto commit = [&](const PrefA& ra, const PrefB& rb, int buf) {
    v8h av;
    #pragma unroll
    for (int u = 0; u < 8; ++u) av[u] = (_Float16)ra.a[u];
    *(v8h*)&As[buf][a_i][a_j8] = av;             // one ds_store_b128

    const float* fb = (const float*)&rb.b[0];
    v8h lo, hi;
    #pragma unroll
    for (int u = 0; u < 8; ++u) { lo[u] = (_Float16)fb[u]; hi[u] = (_Float16)fb[8 + u]; }
    *(v8h*)&Bs[buf][b_r][b_jh]     = lo;
    *(v8h*)&Bs[buf][b_r][b_jh + 8] = hi;
  };

  v8f acc00 = {0.f, 0.f, 0.f, 0.f, 0.f, 0.f, 0.f, 0.f};
  v8f acc01 = acc00, acc10 = acc00, acc11 = acc00;

  const int arow0 = m_sub * 32 + ln16;  // A fragment rows (M = lane%16)
  const int arow1 = arow0 + 16;
  const int c0    = n_sub * 32 + ln16;  // Bs rows for column tiles (N = lane%16)
  const int c1    = c0 + 16;

  // prologue: stage chunk 0 into buffer 0
  {
    PrefA ra = prefetchA(0);
    PrefB rb = prefetchB(0);
    commit(ra, rb, 0);
  }

  #pragma unroll 2
  for (int kk = 0; kk < KTOT; kk += KC) {
    const int p = (kk >> 5) & 1;        // current buffer (constant under unroll 2)
    const int q = p ^ 1;
    const bool have_next = (kk + KC) < KTOT;

    __syncthreads();                    // buf[p] staged; buf[q] free to overwrite

    // issue next chunk's global loads early (hide HBM latency behind WMMA)
    PrefA ra; PrefB rb;
    if (have_next) { ra = prefetchA(kk + KC); rb = prefetchB(kk + KC); }

    // ---- gather fragments per documented wave32 16-bit layouts ----
    // A 16x32: lane half 0 -> K {0..7,16..23}; half 1 -> K {8..15,24..31}
    v8h a0l = *(const v8h*)(&As[p][arow0][hf * 8]);
    v8h a0h = *(const v8h*)(&As[p][arow0][16 + hf * 8]);
    v16h af0 = __builtin_shufflevector(a0l, a0h,
                 0,1,2,3,4,5,6,7,8,9,10,11,12,13,14,15);
    v8h a1l = *(const v8h*)(&As[p][arow1][hf * 8]);
    v8h a1h = *(const v8h*)(&As[p][arow1][16 + hf * 8]);
    v16h af1 = __builtin_shufflevector(a1l, a1h,
                 0,1,2,3,4,5,6,7,8,9,10,11,12,13,14,15);

    // B 32x16: lane = column N; half 0 -> K {0..7,16..23}; half 1 -> K {8..15,24..31}
    v8h b0l = *(const v8h*)(&Bs[p][c0][hf * 8]);
    v8h b0h = *(const v8h*)(&Bs[p][c0][16 + hf * 8]);
    v16h bf0 = __builtin_shufflevector(b0l, b0h,
                 0,1,2,3,4,5,6,7,8,9,10,11,12,13,14,15);
    v8h b1l = *(const v8h*)(&Bs[p][c1][hf * 8]);
    v8h b1h = *(const v8h*)(&Bs[p][c1][16 + hf * 8]);
    v16h bf1 = __builtin_shufflevector(b1l, b1h,
                 0,1,2,3,4,5,6,7,8,9,10,11,12,13,14,15);

    // 4 WMMAs; each fragment feeds two (register reuse, LDS:WMMA = 2:1)
    acc00 = __builtin_amdgcn_wmma_f32_16x16x32_f16(false, af0, false, bf0,
                                                   (short)0, acc00, false, false);
    acc01 = __builtin_amdgcn_wmma_f32_16x16x32_f16(false, af0, false, bf1,
                                                   (short)0, acc01, false, false);
    acc10 = __builtin_amdgcn_wmma_f32_16x16x32_f16(false, af1, false, bf0,
                                                   (short)0, acc10, false, false);
    acc11 = __builtin_amdgcn_wmma_f32_16x16x32_f16(false, af1, false, bf1,
                                                   (short)0, acc11, false, false);

    // commit next chunk into the alternate buffer (read at next iter's barrier)
    if (have_next) commit(ra, rb, q);
  }

  // ---- store: out[b, t*HOP + r]; disjoint across all tiles, coalesced in r ----
  const float scale = 4.0f / (float)NFFT;   // 1/128
  float* ob = out + (size_t)b * LOUT;
  const int r = r0 + n_sub * 32 + ln16;
  #pragma unroll
  for (int v = 0; v < 8; ++v) {
    const int m = v + 8 * hf;               // C/D layout: VGPR v, half-wave -> M
    const int tA = t0 + m_sub * 32 + m;       // m-tile 0
    const int tB = tA + 16;                   // m-tile 1
    if (tA <= T_FRAMES) {                   // valid t in [0, 4096]
      ob[(size_t)tA * HOP + r]      = acc00[v] * scale;
      ob[(size_t)tA * HOP + r + 16] = acc01[v] * scale;
    }
    if (tB <= T_FRAMES) {
      ob[(size_t)tB * HOP + r]      = acc10[v] * scale;
      ob[(size_t)tB * HOP + r + 16] = acc11[v] * scale;
    }
  }
}

extern "C" void kernel_launch(void* const* d_in, const int* in_sizes, int n_in,
                              void* d_out, int out_size, void* d_ws, size_t ws_size,
                              hipStream_t stream) {
  const float* spec    = (const float*)d_in[0];   // [16][1][256][4096] f32
  const float* kernels = (const float*)d_in[1];   // [512][256] f32
  float* out = (float*)d_out;                     // [16][1048832] f32

  dim3 grid(2 /*r blocks*/, 65 /*t blocks: ceil(4097/64)*/, 16 /*batch*/);
  imdct_wmma_kernel<<<grid, dim3(256), 0, stream>>>(spec, kernels, out);
}
